// SingleStreamBlock_86320252715135
// MI455X (gfx1250) — matrix operand order repaired and checked by reading, not compiled, not run on hardware
//
#include <hip/hip_runtime.h>
#include <hip/hip_bf16.h>
#include <math.h>

// ---------------------------------------------------------------------------
// Types
// ---------------------------------------------------------------------------
typedef __bf16 bf16_t;
typedef __attribute__((ext_vector_type(16))) __bf16 v16bf;
typedef __attribute__((ext_vector_type(8)))  __bf16 v8bf;
typedef __attribute__((ext_vector_type(8)))  float  v8f;
typedef __attribute__((ext_vector_type(4)))  int    v4i;
typedef __attribute__((ext_vector_type(4)))  unsigned u32x4;
typedef __attribute__((ext_vector_type(8)))  int    i32x8;
typedef __attribute__((ext_vector_type(4)))  int    i32x4;

// Problem constants
#define D      3072
#define H      24
#define DH     128
#define MLP    12288
#define L      2048
#define NQKV   9216              // 3*D
#define N1     21504             // 3*D + MLP
#define N2CAT  15360             // D + MLP  (concat width, K of gemm2)

// gfx1250 async global->LDS path (ASYNCcnt-tracked), with safe fallback.
#if defined(__gfx1250__) && __has_builtin(__builtin_amdgcn_global_load_async_to_lds_b128)
#define HAVE_ASYNC_LDS 1
#else
#define HAVE_ASYNC_LDS 0
#endif

// gfx1250 Tensor Data Mover (TENSORcnt-tracked); this toolchain's builtin is
// the 6-arg form: (u32x4 g0, i32x8 g1, i32x4 g2, i32x4 g3, i32x8 gx, i32 cpol).
#if defined(__gfx1250__) && __has_builtin(__builtin_amdgcn_tensor_load_to_lds) && \
    __has_builtin(__builtin_amdgcn_s_wait_tensorcnt)
#define HAVE_TDM 1
#else
#define HAVE_TDM 0
#endif

// ---------------------------------------------------------------------------
// Helpers
// ---------------------------------------------------------------------------
__device__ __forceinline__ bf16_t f2bf(float f) {
  unsigned u = __builtin_bit_cast(unsigned, f);
  u += 0x7FFFu + ((u >> 16) & 1u);          // round-to-nearest-even
  unsigned short h = (unsigned short)(u >> 16);
  return __builtin_bit_cast(bf16_t, h);
}

__device__ __forceinline__ v8f vzero8() {
  v8f z;
#pragma unroll
  for (int i = 0; i < 8; ++i) z[i] = 0.0f;
  return z;
}

// A-fragment (16x32 bf16, MxK): lane holds M=lane%16; K runs of 8:
// lanes 0-15 -> K {0..7, 16..23}, lanes 16-31 -> K {8..15, 24..31}
__device__ __forceinline__ v16bf load_frag_a(const bf16_t* rowk, int hi) {
  const v8bf* p0 = (const v8bf*)(rowk + (hi ? 8 : 0));
  const v8bf* p1 = (const v8bf*)(rowk + 16 + (hi ? 8 : 0));
  v8bf lo = *p0;
  v8bf hh = *p1;
  v16bf f;
#pragma unroll
  for (int i = 0; i < 8; ++i) { f[i] = lo[i]; f[8 + i] = hh[i]; }
  return f;
}

// B-fragment (32x16 bf16, KxN): lane holds N=lane%16; K runs of 8:
// lanes 0-15 -> K {0..7, 16..23}, lanes 16-31 -> K {8..15, 24..31}
__device__ __forceinline__ v16bf load_frag_b(const bf16_t* rowk, int hi) {
  const v8bf* p = (const v8bf*)(rowk + (hi ? 8 : 0));
  v8bf lo = p[0];
  v8bf hh = p[2];
  v16bf f;
#pragma unroll
  for (int i = 0; i < 8; ++i) { f[i] = lo[i]; f[8 + i] = hh[i]; }
  return f;
}

// One 16B chunk: global -> LDS.  Async (ASYNCcnt) on gfx1250 if available.
__device__ __forceinline__ void stage16(const bf16_t* g, bf16_t* l) {
#if HAVE_ASYNC_LDS
  v4i* gp = (v4i*)(void*)g;   // drop const, generic typed
  v4i* lp = (v4i*)(void*)l;
  __builtin_amdgcn_global_load_async_to_lds_b128(
      (__attribute__((address_space(1))) v4i*)gp,
      (__attribute__((address_space(3))) v4i*)lp, 0, 0);
#else
  *(v8bf*)l = *(const v8bf*)g;
#endif
}

#if HAVE_TDM
// One TDM descriptor: 2D tile, 64 rows x 64 bf16 (128B/row), row stride =
// Kdim elements, global -> LDS contiguous row-major.  D# layout per ISA
// "Tensor DMA Descriptor" tables (group0: count/lds_addr/global_addr/type,
// group1: data_size + tensor/tile dims + dim0 stride; groups 2/3 unused).
__device__ __forceinline__ void tdm_stage_tile(const bf16_t* g, bf16_t* l, int Kdim) {
  unsigned lds_addr =
      (unsigned)(size_t)(__attribute__((address_space(3))) bf16_t*)l;
  unsigned long long ga = (unsigned long long)(size_t)g;
  u32x4 g0;
  g0[0] = 1u;                                              // count=1 (user D#)
  g0[1] = lds_addr;                                        // lds_addr[31:0]
  g0[2] = (unsigned)(ga & 0xFFFFFFFFu);                    // global_addr[31:0]
  g0[3] = (unsigned)((ga >> 32) & 0x01FFFFFFu)             // global_addr[56:32]
        | 0x80000000u;                                     // type=2 ("image")
  i32x8 g1;
  g1[0] = 0x00010000;                                      // data_size=1 (2B)
  g1[1] = 64 << 16;                                        // tensor_dim0=64 (lo16)
  g1[2] = 64 << 16;                                        // td0 hi=0 | tensor_dim1=64
  g1[3] = 64 << 16;                                        // td1 hi=0 | tile_dim0=64
  g1[4] = 64;                                              // tile_dim1=64, tile_dim2=0
  g1[5] = Kdim;                                            // tensor_dim0_stride[31:0]
  g1[6] = 0;                                               // stride hi | dim1_stride lo
  g1[7] = 0;
  i32x4 z4 = {0, 0, 0, 0};
  i32x8 z8 = {0, 0, 0, 0, 0, 0, 0, 0};
  __builtin_amdgcn_tensor_load_to_lds(g0, g1, z4, z4, z8, 0);
}
#endif

__device__ __forceinline__ void wait_stage() {
#if HAVE_TDM
  __builtin_amdgcn_s_wait_tensorcnt(0);
#elif HAVE_ASYNC_LDS
#if __has_builtin(__builtin_amdgcn_s_wait_asynccnt)
  __builtin_amdgcn_s_wait_asynccnt(0);
#else
  asm volatile("s_wait_asynccnt 0x0" ::: "memory");
#endif
#endif
}

__device__ __forceinline__ float gelu_tanh(float v) {
  return 0.5f * v * (1.0f + tanhf(0.79788456080286536f * (v + 0.044715f * v * v * v)));
}

// ---------------------------------------------------------------------------
// Shared WMMA GEMM core: block = 256 thr (8 waves), tile M=128 x N=64, BK=64.
// B-tile (64 rows x 64 k, bf16 = 8KB) double-buffered in LDS, staged by the
// Tensor Data Mover (one descriptor per tile, wave 0) or per-lane async
// loads; prefetch for step i+1 is issued before the WMMAs of step i.
// B-fragments are pre-loaded in groups of 4 so WMMAs issue back-to-back;
// the A-fragment for the next substep is prefetched from global early.
// ---------------------------------------------------------------------------
__device__ __forceinline__ void wmma_gemm_core(const bf16_t* __restrict__ A,
                                               const bf16_t* __restrict__ W,
                                               int Kdim, int n0, int m0wave,
                                               bf16_t Bs[2][64][64],
                                               v8f acc[4]) {
  const int tid  = threadIdx.x;
  const int lane = tid & 31;
  const int wave = tid >> 5;
  const int hi   = lane >> 4;
  const int lr   = lane & 15;
  const int row0 = tid >> 3;            // 0..31
  const int sub  = tid & 7;             // 0..7  (16B chunk within 128B row)

#pragma unroll
  for (int t = 0; t < 4; ++t) acc[t] = vzero8();

  const bf16_t* aRow = A + (size_t)(m0wave + lr) * Kdim;

  // prologue: stage first tile into buffer 0
#if HAVE_TDM
  if (wave == 0) tdm_stage_tile(W + (size_t)n0 * Kdim, &Bs[0][0][0], Kdim);
#else
  stage16(W + (size_t)(n0 + row0)      * Kdim + sub * 8, &Bs[0][row0][sub * 8]);
  stage16(W + (size_t)(n0 + row0 + 32) * Kdim + sub * 8, &Bs[0][row0 + 32][sub * 8]);
#endif
  wait_stage();
  __syncthreads();

  v16bf a_cur = load_frag_a(aRow, hi);
  int cur = 0;
  for (int k = 0; k < Kdim; k += 64) {
    const int last = (k + 64 >= Kdim);
    // prefetch next tile into the other buffer (hidden behind WMMAs)
    if (!last) {
#if HAVE_TDM
      if (wave == 0)
        tdm_stage_tile(W + (size_t)n0 * Kdim + k + 64, &Bs[cur ^ 1][0][0], Kdim);
#else
      stage16(W + (size_t)(n0 + row0)      * Kdim + k + 64 + sub * 8,
              &Bs[cur ^ 1][row0][sub * 8]);
      stage16(W + (size_t)(n0 + row0 + 32) * Kdim + k + 64 + sub * 8,
              &Bs[cur ^ 1][row0 + 32][sub * 8]);
#endif
    }
    // substep 0: prefetch substep-1 A, gather 4 B frags, 4 WMMAs back-to-back
    v16bf a_mid = load_frag_a(aRow + k + 32, hi);
    v16bf b0[4];
#pragma unroll
    for (int t = 0; t < 4; ++t) b0[t] = load_frag_b(&Bs[cur][t * 16 + lr][0], hi);
#pragma unroll
    for (int t = 0; t < 4; ++t)
      acc[t] = __builtin_amdgcn_wmma_f32_16x16x32_bf16(false, a_cur, false, b0[t],
                                                       (short)0, acc[t], false, false);
    // substep 1: prefetch next-iter A (dummy re-read of k=0 on last iter)
    v16bf a_nxt = load_frag_a(aRow + (last ? 0 : k + 64), hi);
    v16bf b1[4];
#pragma unroll
    for (int t = 0; t < 4; ++t) b1[t] = load_frag_b(&Bs[cur][t * 16 + lr][32], hi);
#pragma unroll
    for (int t = 0; t < 4; ++t)
      acc[t] = __builtin_amdgcn_wmma_f32_16x16x32_bf16(false, a_mid, false, b1[t],
                                                       (short)0, acc[t], false, false);
    a_cur = a_nxt;
    wait_stage();
    __syncthreads();
    cur ^= 1;
  }
}

// ---------------------------------------------------------------------------
// Kernel: fp32 -> bf16 weight conversion (grid-stride)
// ---------------------------------------------------------------------------
__global__ void k_f32_to_bf16(const float* __restrict__ in, bf16_t* __restrict__ out, long n) {
  long i = (long)blockIdx.x * blockDim.x + threadIdx.x;
  long stride = (long)gridDim.x * blockDim.x;
  for (; i < n; i += stride) out[i] = f2bf(in[i]);
}

// ---------------------------------------------------------------------------
// Kernel: silu(vec)
// ---------------------------------------------------------------------------
__global__ void k_silu(const float* __restrict__ vec, float* __restrict__ sv) {
  int i = blockIdx.x * blockDim.x + threadIdx.x;
  if (i < D) {
    float v = vec[i];
    sv[i] = v / (1.0f + __expf(-v));
  }
}

// ---------------------------------------------------------------------------
// Kernel: mod = silu(vec) @ mod_w.T + mod_b   (GEMV, 9216 outputs, K=3072)
// ---------------------------------------------------------------------------
__global__ __launch_bounds__(256) void k_modvec(const float* __restrict__ sv,
                                                const float* __restrict__ mw,
                                                const float* __restrict__ mb,
                                                float* __restrict__ mod) {
  int j = blockIdx.x * 256 + threadIdx.x;
  if (j >= 3 * D) return;
  const float* r = mw + (size_t)j * D;
  float acc = 0.0f;
  for (int d = 0; d < D; ++d) acc = fmaf(sv[d], r[d], acc);
  mod[j] = acc + mb[j];
}

// ---------------------------------------------------------------------------
// Kernel: x_mod = (1+scale)*LN(x) + shift, written as bf16 (one block per row)
// ---------------------------------------------------------------------------
__global__ __launch_bounds__(256) void k_ln_mod(const float* __restrict__ x,
                                                const float* __restrict__ mod,
                                                bf16_t* __restrict__ xmodb) {
  const int m = blockIdx.x;
  const float* row = x + (size_t)m * D;
  float s = 0.0f, s2 = 0.0f;
  for (int c = threadIdx.x; c < D; c += 256) {
    float v = row[c];
    s += v; s2 += v * v;
  }
#pragma unroll
  for (int off = 1; off < 32; off <<= 1) {
    s  += __shfl_xor(s,  off, 32);
    s2 += __shfl_xor(s2, off, 32);
  }
  __shared__ float sh[2][8];
  const int lane = threadIdx.x & 31, wv = threadIdx.x >> 5;
  if (lane == 0) { sh[0][wv] = s; sh[1][wv] = s2; }
  __syncthreads();
  if (threadIdx.x < 32) {
    float a = (lane < 8) ? sh[0][lane] : 0.0f;
    float b = (lane < 8) ? sh[1][lane] : 0.0f;
#pragma unroll
    for (int off = 1; off < 8; off <<= 1) {
      a += __shfl_xor(a, off, 32);
      b += __shfl_xor(b, off, 32);
    }
    if (lane == 0) { sh[0][0] = a; sh[1][0] = b; }
  }
  __syncthreads();
  const float mu  = sh[0][0] * (1.0f / D);
  const float ex2 = sh[1][0] * (1.0f / D);
  const float rstd = rsqrtf(ex2 - mu * mu + 1e-6f);
  for (int c = threadIdx.x; c < D; c += 256) {
    float v = (row[c] - mu) * rstd;
    xmodb[(size_t)m * D + c] = f2bf((1.0f + mod[D + c]) * v + mod[c]);
  }
}

// ---------------------------------------------------------------------------
// Kernel: GEMM1  h = x_mod @ w1.T + b1   (M=2048, N=21504, K=3072, bf16 WMMA)
// Fused epilogue: q/k -> fp32 [head][seq][dh]; v -> bf16 transposed
// [head][dh][seq]; mlp -> gelu -> bf16 concat buffer cols [3072, 15360).
// ---------------------------------------------------------------------------
__global__ __launch_bounds__(256) void k_gemm1(const bf16_t* __restrict__ A,
                                               const bf16_t* __restrict__ W,
                                               const float*  __restrict__ b1,
                                               float* __restrict__ qf,
                                               float* __restrict__ kf,
                                               bf16_t* __restrict__ vbT,
                                               bf16_t* __restrict__ concatb) {
  __shared__ __attribute__((aligned(16))) bf16_t Bs[2][64][64];
  const int lane = threadIdx.x & 31;
  const int wave = threadIdx.x >> 5;
  const int hi   = lane >> 4;
  const int lr   = lane & 15;
  const int n0   = blockIdx.x * 64;
  const int m0   = blockIdx.y * 128 + wave * 16;

  v8f acc[4];
  wmma_gemm_core(A, W, D, n0, m0, Bs, acc);

#pragma unroll
  for (int t = 0; t < 4; ++t) {
    const int n = n0 + t * 16 + lr;
    const float bias = b1[n];
#pragma unroll
    for (int r = 0; r < 8; ++r) {
      const int m = m0 + r + hi * 8;
      const float v = acc[t][r] + bias;
      if (n < NQKV) {
        const int part = n / D;
        const int rem  = n % D;
        const int head = rem >> 7;
        const int dh   = rem & 127;
        const size_t idx = (((size_t)head * L) + m) * DH + dh;
        if (part == 0)      qf[idx] = v;
        else if (part == 1) kf[idx] = v;
        else vbT[(((size_t)head * DH) + dh) * L + m] = f2bf(v);
      } else {
        concatb[(size_t)m * N2CAT + D + (n - NQKV)] = f2bf(gelu_tanh(v));
      }
    }
  }
}

// ---------------------------------------------------------------------------
// Kernel: RMSNorm + RoPE for q or k. One wave per (head,seq) row of 128.
// in/out layout: [head][seq][dh].  pe layout: [seq][64][2][2].
// ---------------------------------------------------------------------------
__global__ __launch_bounds__(256) void k_qk_norm_rope(const float* __restrict__ in,
                                                      const float* __restrict__ scale,
                                                      const float* __restrict__ pe,
                                                      bf16_t* __restrict__ outb) {
  const int lane = threadIdx.x & 31;
  const int wv   = threadIdx.x >> 5;
  const int wid  = blockIdx.x * 8 + wv;           // head*L + seq
  const int seq  = wid & (L - 1);

  const float* row = in + (size_t)wid * DH;
  float x0 = row[lane * 4 + 0];
  float x1 = row[lane * 4 + 1];
  float x2 = row[lane * 4 + 2];
  float x3 = row[lane * 4 + 3];

  float ss = x0 * x0 + x1 * x1 + x2 * x2 + x3 * x3;
#pragma unroll
  for (int off = 1; off < 32; off <<= 1) ss += __shfl_xor(ss, off, 32);
  const float rr = rsqrtf(ss * (1.0f / DH) + 1e-6f);

  x0 *= rr * scale[lane * 4 + 0];
  x1 *= rr * scale[lane * 4 + 1];
  x2 *= rr * scale[lane * 4 + 2];
  x3 *= rr * scale[lane * 4 + 3];

  const float* p = pe + (((size_t)seq * 64) + lane * 2) * 4;  // two pairs per lane
  float o0 = p[0] * x0 + p[1] * x1;
  float o1 = p[2] * x0 + p[3] * x1;
  float o2 = p[4] * x2 + p[5] * x3;
  float o3 = p[6] * x2 + p[7] * x3;

  bf16_t* o = outb + (size_t)wid * DH + lane * 4;
  o[0] = f2bf(o0); o[1] = f2bf(o1); o[2] = f2bf(o2); o[3] = f2bf(o3);
}

// ---------------------------------------------------------------------------
// Kernel: flash attention. grid(L/128, H); 8 waves/block; one wave per
// 16-row Q tile. QK^T and P.V via bf16 WMMA; P staged through LDS to
// convert C-layout -> A-fragment layout. Output to concat cols [0,3072).
// ---------------------------------------------------------------------------
__global__ __launch_bounds__(256) void k_attn(const bf16_t* __restrict__ qb,
                                              const bf16_t* __restrict__ kb,
                                              const bf16_t* __restrict__ vbT,
                                              bf16_t* __restrict__ concatb) {
  __shared__ __attribute__((aligned(16))) bf16_t P[8][16][32];

  const int lane = threadIdx.x & 31;
  const int wave = threadIdx.x >> 5;
  const int hi   = lane >> 4;
  const int lr   = lane & 15;
  const int head = blockIdx.y;
  const int q0   = blockIdx.x * 128 + wave * 16;

  // Q tile fragments for the whole 16x128 tile (4 K-slices of 32)
  const bf16_t* qBase = qb + ((size_t)head * L + q0 + lr) * DH;
  v16bf qfr[4];
#pragma unroll
  for (int kk = 0; kk < 4; ++kk) qfr[kk] = load_frag_a(qBase + kk * 32, hi);

  const bf16_t* kBase = kb  + ((size_t)head * L  + lr) * DH;
  const bf16_t* vBase = vbT + ((size_t)head * DH + lr) * L;

  float mrow[8], lrow[8];
  v8f o[8];
#pragma unroll
  for (int r = 0; r < 8; ++r) { mrow[r] = -1e30f; lrow[r] = 0.0f; }
#pragma unroll
  for (int dt = 0; dt < 8; ++dt) o[dt] = vzero8();

  const float inv_scale = 0.08838834764831845f;   // DH^-0.5

  for (int kt = 0; kt < L; kt += 32) {
    // S = Q @ K^T for 16x32 keys (two 16x16 accumulators)
    v8f s0 = vzero8(), s1 = vzero8();
#pragma unroll
    for (int kk = 0; kk < 4; ++kk) {
      v16bf b0 = load_frag_b(kBase + (size_t)(kt)      * DH + kk * 32, hi);
      v16bf b1 = load_frag_b(kBase + (size_t)(kt + 16) * DH + kk * 32, hi);
      s0 = __builtin_amdgcn_wmma_f32_16x16x32_bf16(false, qfr[kk], false, b0,
                                                   (short)0, s0, false, false);
      s1 = __builtin_amdgcn_wmma_f32_16x16x32_bf16(false, qfr[kk], false, b1,
                                                   (short)0, s1, false, false);
    }

    // Online softmax per row (row = r + hi*8; cols striped over 16 lanes/half)
#pragma unroll
    for (int r = 0; r < 8; ++r) {
      float v0 = s0[r] * inv_scale;
      float v1 = s1[r] * inv_scale;
      float mx = fmaxf(v0, v1);
#pragma unroll
      for (int off = 1; off < 16; off <<= 1) mx = fmaxf(mx, __shfl_xor(mx, off, 32));
      const float mnew  = fmaxf(mrow[r], mx);
      const float alpha = __expf(mrow[r] - mnew);
      const float p0 = __expf(v0 - mnew);
      const float p1 = __expf(v1 - mnew);
      float ps = p0 + p1;
#pragma unroll
      for (int off = 1; off < 16; off <<= 1) ps += __shfl_xor(ps, off, 32);
      lrow[r] = lrow[r] * alpha + ps;
      mrow[r] = mnew;
#pragma unroll
      for (int dt = 0; dt < 8; ++dt) o[dt][r] *= alpha;
      P[wave][r + hi * 8][lr]      = f2bf(p0);
      P[wave][r + hi * 8][16 + lr] = f2bf(p1);
    }
    __syncthreads();

    // O += P(16x32) @ V(32x128): reload P as an A-fragment from LDS
    v16bf pf = load_frag_a(&P[wave][lr][0], hi);
#pragma unroll
    for (int dt = 0; dt < 8; ++dt) {
      v16bf vf = load_frag_b(vBase + (size_t)dt * 16 * L + kt, hi);
      o[dt] = __builtin_amdgcn_wmma_f32_16x16x32_bf16(false, pf, false, vf,
                                                      (short)0, o[dt], false, false);
    }
    __syncthreads();
  }

#pragma unroll
  for (int r = 0; r < 8; ++r) {
    const float il = 1.0f / lrow[r];
    const int m = q0 + r + hi * 8;
#pragma unroll
    for (int dt = 0; dt < 8; ++dt) {
      concatb[(size_t)m * N2CAT + head * DH + dt * 16 + lr] = f2bf(o[dt][r] * il);
    }
  }
}

// ---------------------------------------------------------------------------
// Kernel: GEMM2  out = x + gate * (concat @ w2.T + b2)
// (M=2048, N=3072, K=15360, bf16 WMMA, fused residual epilogue)
// ---------------------------------------------------------------------------
__global__ __launch_bounds__(256) void k_gemm2(const bf16_t* __restrict__ A,
                                               const bf16_t* __restrict__ W,
                                               const float*  __restrict__ b2,
                                               const float*  __restrict__ x,
                                               const float*  __restrict__ mod,
                                               float* __restrict__ out) {
  __shared__ __attribute__((aligned(16))) bf16_t Bs[2][64][64];
  const int lane = threadIdx.x & 31;
  const int wave = threadIdx.x >> 5;
  const int hi   = lane >> 4;
  const int lr   = lane & 15;
  const int n0   = blockIdx.x * 64;
  const int m0   = blockIdx.y * 128 + wave * 16;

  v8f acc[4];
  wmma_gemm_core(A, W, N2CAT, n0, m0, Bs, acc);

#pragma unroll
  for (int t = 0; t < 4; ++t) {
    const int n = n0 + t * 16 + lr;
    const float bias = b2[n];
    const float gate = mod[2 * D + n];
#pragma unroll
    for (int r = 0; r < 8; ++r) {
      const int m = m0 + r + hi * 8;
      const size_t idx = (size_t)m * D + n;
      out[idx] = x[idx] + gate * (acc[t][r] + bias);
    }
  }
}

// ---------------------------------------------------------------------------
// Host launcher
// ---------------------------------------------------------------------------
extern "C" void kernel_launch(void* const* d_in, const int* in_sizes, int n_in,
                              void* d_out, int out_size, void* d_ws, size_t ws_size,
                              hipStream_t stream) {
  const float* x       = (const float*)d_in[0];
  const float* vec     = (const float*)d_in[1];
  const float* pe      = (const float*)d_in[2];
  const float* mod_w   = (const float*)d_in[3];
  const float* mod_b   = (const float*)d_in[4];
  const float* w1      = (const float*)d_in[5];
  const float* b1      = (const float*)d_in[6];
  const float* w2      = (const float*)d_in[7];
  const float* b2      = (const float*)d_in[8];
  const float* q_scale = (const float*)d_in[9];
  const float* k_scale = (const float*)d_in[10];
  float* out = (float*)d_out;

  // Workspace layout (256B aligned slabs)
  char* ws = (char*)d_ws;
  size_t off = 0;
  auto alloc = [&](size_t bytes) -> void* {
    void* p = ws + off;
    off += (bytes + 255) & ~(size_t)255;
    return p;
  };
  float*  sv      = (float*) alloc((size_t)D * 4);                 // silu(vec)
  float*  mod     = (float*) alloc((size_t)3 * D * 4);             // shift|scale|gate
  bf16_t* w1b     = (bf16_t*)alloc((size_t)N1 * D * 2);            // bf16 w1
  bf16_t* w2b     = (bf16_t*)alloc((size_t)D * N2CAT * 2);         // bf16 w2
  bf16_t* xmodb   = (bf16_t*)alloc((size_t)L * D * 2);             // modulated LN(x)
  float*  qf      = (float*) alloc((size_t)H * L * DH * 4);        // q fp32
  float*  kf      = (float*) alloc((size_t)H * L * DH * 4);        // k fp32
  bf16_t* qbn     = (bf16_t*)alloc((size_t)H * L * DH * 2);        // q post norm+rope
  bf16_t* kbn     = (bf16_t*)alloc((size_t)H * L * DH * 2);        // k post norm+rope
  bf16_t* vbT     = (bf16_t*)alloc((size_t)H * DH * L * 2);        // v transposed
  bf16_t* concatb = (bf16_t*)alloc((size_t)L * N2CAT * 2);         // [attn | gelu(mlp)]

  // 1) Weights -> bf16 (each then fits in the 192MB L2 for the GEMM K-loops)
  k_f32_to_bf16<<<8192, 256, 0, stream>>>(w1, w1b, (long)N1 * D);
  k_f32_to_bf16<<<8192, 256, 0, stream>>>(w2, w2b, (long)D * N2CAT);

  // 2) mod vector
  k_silu<<<(D + 255) / 256, 256, 0, stream>>>(vec, sv);
  k_modvec<<<(3 * D + 255) / 256, 256, 0, stream>>>(sv, mod_w, mod_b, mod);

  // 3) LN + modulate -> bf16 A for GEMM1
  k_ln_mod<<<L, 256, 0, stream>>>(x, mod, xmodb);

  // 4) GEMM1 + fused qkv-split / gelu epilogue
  {
    dim3 grid(N1 / 64, L / 128);
    k_gemm1<<<grid, 256, 0, stream>>>(xmodb, w1b, b1, qf, kf, vbT, concatb);
  }

  // 5) RMSNorm + RoPE on q and k
  k_qk_norm_rope<<<(H * L) / 8, 256, 0, stream>>>(qf, q_scale, pe, qbn);
  k_qk_norm_rope<<<(H * L) / 8, 256, 0, stream>>>(kf, k_scale, pe, kbn);

  // 6) Flash attention -> concat cols [0, 3072)
  {
    dim3 grid(L / 128, H);
    k_attn<<<grid, 256, 0, stream>>>(qbn, kbn, vbT, concatb);
  }

  // 7) GEMM2 + fused residual
  {
    dim3 grid(D / 64, L / 128);
    k_gemm2<<<grid, 256, 0, stream>>>(concatb, w2b, b2, x, mod, out);
  }

  (void)in_sizes; (void)n_in; (void)out_size; (void)ws_size;
}
